// ForcePai_39462159515768
// MI455X (gfx1250) — matrix operand order repaired and checked by reading, not compiled, not run on hardware
//
// CDNA5 / gfx1250 implementation of the PaiNN-style "ForcePai" reference.
// Round 4: explicit double-buffered software pipeline in the WMMA K-loop —
// fragments of step k+1 are loaded before the WMMAs of step k, forcing the
// allocator to keep cur/next in distinct registers so WMMA execution overlaps
// the outstanding loads (removes the per-WMMA s_wait_loadcnt 0x0 seen in
// round 3 steady state). Everything else unchanged.
//
// Input pointer layout (JAX pytree flatten, dict keys ASCII-sorted):
//   d_in[0] = nxyz (10000x4 f32), d_in[1] = nbrs (160000x2 i32),
//   d_in[2] = params['embed'] (100x128 f32),
//   d_in[3 + L*26 + k] = layer L param k (alphabetical order, see enum).
// Workspace requirement: ~710 MB.

#include <hip/hip_runtime.h>
#include <hip/hip_bf16.h>
#include <math.h>

typedef _Float16 half_t;
typedef __attribute__((ext_vector_type(8)))  _Float16 v8h;
typedef __attribute__((ext_vector_type(16))) _Float16 v16h;
typedef __attribute__((ext_vector_type(8)))  float    v8f;

#define NATOMS 10000
#define NEDGES 160000
#define FEAT   128
#define NRBF   20
#define RBFPAD 32
#define CUT    5.0f
#define EPS3   (3.0f * 1e-15f)
#define PI_F   3.14159265358979323846f

#define CEILDIV(a, b) (((a) + (b) - 1) / (b))

__device__ __forceinline__ float silu_f(float x) {
  return x * __builtin_amdgcn_rcpf(1.0f + __expf(-x));
}

__device__ __forceinline__ void atomAddF(float* p, float v) {
  __hip_atomic_fetch_add(p, v, __ATOMIC_RELAXED, __HIP_MEMORY_SCOPE_AGENT);
}

// ---------------------------------------------------------------------------
// WMMA GEMM: C[M,N] = epilogue(A[M,K] @ B[K,N])
//   A f16 row-major; B passed TRANSPOSED as Bt[N,K] f16 row-major.
//   Wave computes a 16x64 strip (4 N-tiles share one A fragment), with the
//   K-loop explicitly double-buffered for load/WMMA overlap.
//   Fragment lane layouts per CDNA5 ISA 7.12.2 (wave32):
//     A 16x32: lane(hh,mr) holds row mr, halfs [k0+8hh..+7] and [k0+16+8hh..+7]
//     B 32x16: lane(hh,mr) holds col mr, halfs [k0+8hh..+7] and [k0+16+8hh..+7]
//     C 16x16: vgpr r -> row r+8hh, col mr
// ---------------------------------------------------------------------------
enum { GF_BIAS = 1, GF_SILU = 2, GF_RS = 4, GF_OF32 = 8, GF_OF16 = 16 };

template <int K, int FLAGS>
__global__ __launch_bounds__(256) void gemm_wmma(
    const half_t* __restrict__ A, const half_t* __restrict__ Bt,
    const float* __restrict__ bias, const float* __restrict__ rowScale,
    float* __restrict__ Cf, half_t* __restrict__ Ch, int M, int N)
{
  constexpr int NT = 4;                       // N-tiles per wave (64 cols)
  int groups = N >> 6;
  int tiles  = (M >> 4) * groups;
  int tile   = (int)blockIdx.x * (blockDim.x >> 5) + ((int)threadIdx.x >> 5);
  if (tile >= tiles) return;                  // whole-wave exit: EXEC stays all-1s
  int tm = tile / groups, tg = tile % groups;
  int lane = threadIdx.x & 31;
  int hh = lane >> 4, mr = lane & 15;

  const half_t* arow = A + (size_t)(tm * 16 + mr) * K + 8 * hh;
  const half_t* brow[NT];
#pragma unroll
  for (int nt = 0; nt < NT; ++nt)
    brow[nt] = Bt + (size_t)((tg * NT + nt) * 16 + mr) * K + 8 * hh;

  auto loadA = [&](int k0) -> v16h {
    v8h lo = *(const v8h*)(arow + k0);        // global_load_b128
    v8h hi = *(const v8h*)(arow + k0 + 16);   // global_load_b128
    return __builtin_shufflevector(lo, hi,
        0, 1, 2, 3, 4, 5, 6, 7, 8, 9, 10, 11, 12, 13, 14, 15);
  };
  auto loadB = [&](int nt, int k0) -> v16h {
    v8h lo = *(const v8h*)(brow[nt] + k0);
    v8h hi = *(const v8h*)(brow[nt] + k0 + 16);
    return __builtin_shufflevector(lo, hi,
        0, 1, 2, 3, 4, 5, 6, 7, 8, 9, 10, 11, 12, 13, 14, 15);
  };

  v8f acc[NT] = {};
  // prologue: fragments for k0 = 0
  v16h af = loadA(0);
  v16h bf[NT];
#pragma unroll
  for (int nt = 0; nt < NT; ++nt) bf[nt] = loadB(nt, 0);

#pragma unroll
  for (int k0 = 0; k0 < K; k0 += 32) {
    // issue next step's loads before this step's WMMAs (software pipeline)
    v16h af_n = af;
    v16h bf_n[NT];
#pragma unroll
    for (int nt = 0; nt < NT; ++nt) bf_n[nt] = bf[nt];
    if (k0 + 32 < K) {
      af_n = loadA(k0 + 32);
#pragma unroll
      for (int nt = 0; nt < NT; ++nt) bf_n[nt] = loadB(nt, k0 + 32);
    }
#pragma unroll
    for (int nt = 0; nt < NT; ++nt)
      acc[nt] = __builtin_amdgcn_wmma_f32_16x16x32_f16(false, af, false, bf[nt],
                                                       (short)0, acc[nt], false, false);
    af = af_n;
#pragma unroll
    for (int nt = 0; nt < NT; ++nt) bf[nt] = bf_n[nt];
  }

  float bv[NT];
#pragma unroll
  for (int nt = 0; nt < NT; ++nt)
    bv[nt] = (FLAGS & GF_BIAS) ? bias[(tg * NT + nt) * 16 + mr] : 0.0f;

#pragma unroll
  for (int r = 0; r < 8; ++r) {
    int row = tm * 16 + r + 8 * hh;
    float rsv = (FLAGS & GF_RS) ? rowScale[row] : 1.0f;
#pragma unroll
    for (int nt = 0; nt < NT; ++nt) {
      int col = (tg * NT + nt) * 16 + mr;
      float cv = acc[nt][r] + bv[nt];
      if constexpr (FLAGS & GF_SILU) cv = silu_f(cv);
      if constexpr (FLAGS & GF_RS)   cv *= rsv;
      size_t o = (size_t)row * N + col;
      if constexpr (FLAGS & GF_OF32) Cf[o] = cv;
      if constexpr (FLAGS & GF_OF16) Ch[o] = (half_t)cv;
    }
  }
}

template <int K, int FLAGS>
static inline void gemm_t(hipStream_t st, const half_t* A, const half_t* Bt,
                          const float* bias, const float* rs,
                          float* Cf, half_t* Ch, int M, int N) {
  int tiles = (M / 16) * (N / 64);
  gemm_wmma<K, FLAGS><<<CEILDIV(tiles, 8), 256, 0, st>>>(A, Bt, bias, rs, Cf, Ch, M, N);
}

// ---------------------------------------------------------------------------
// Elementwise / gather / scatter kernels
// ---------------------------------------------------------------------------
__global__ void k_zero(float* p, int n) {
  int i = blockIdx.x * blockDim.x + threadIdx.x;
  if (i < n) p[i] = 0.0f;
}

__global__ void k_f2h(half_t* d, const float* s, int n) {
  int i = blockIdx.x * blockDim.x + threadIdx.x;
  if (i < n) d[i] = (half_t)s[i];
}

// f32 W[K,N] -> f16 TRANSPOSED Bt[N,Kpad] with K-padding (k >= Kvalid zeroed)
__global__ void k_wpadT(half_t* d, const float* s, int Kvalid, int Kpad, int N) {
  int i = blockIdx.x * blockDim.x + threadIdx.x;
  if (i >= Kpad * N) return;
  int n = i / Kpad, k = i - n * Kpad;
  d[i] = (k < Kvalid) ? (half_t)s[k * N + n] : (half_t)0.0f;
}

__global__ void k_embed(const float* __restrict__ nxyz, const float* __restrict__ emb,
                        float* __restrict__ s, half_t* __restrict__ sh) {
  int i = blockIdx.x * blockDim.x + threadIdx.x;
  if (i >= NATOMS * FEAT) return;
  int n = i >> 7, f = i & 127;
  int z = (int)nxyz[n * 4];
  float v = emb[z * FEAT + f];
  s[i] = v; sh[i] = (half_t)v;
}

__global__ void k_geom(const float* __restrict__ nxyz, const int* __restrict__ nbrs,
                       float* __restrict__ unit, float* __restrict__ env,
                       half_t* __restrict__ erbf) {
  int e = blockIdx.x * blockDim.x + threadIdx.x;
  if (e >= NEDGES) return;
  int n0 = nbrs[2 * e], n1 = nbrs[2 * e + 1];
  float rx = nxyz[n1 * 4 + 1] - nxyz[n0 * 4 + 1];
  float ry = nxyz[n1 * 4 + 2] - nxyz[n0 * 4 + 2];
  float rz = nxyz[n1 * 4 + 3] - nxyz[n0 * 4 + 3];
  float d = sqrtf(rx * rx + ry * ry + rz * rz + EPS3);
  float inv = 1.0f / d;
  unit[e * 3 + 0] = rx * inv; unit[e * 3 + 1] = ry * inv; unit[e * 3 + 2] = rz * inv;
  env[e] = (d < CUT) ? 0.5f * (__cosf(PI_F * d / CUT) + 1.0f) : 0.0f;
  float coef = PI_F / CUT;
  half_t* er = erbf + (size_t)e * RBFPAD;
#pragma unroll
  for (int n = 0; n < NRBF; ++n)
    er[n] = (half_t)((d >= CUT) ? 0.0f : __sinf((float)(n + 1) * coef * d) * inv);
#pragma unroll
  for (int n = NRBF; n < RBFPAD; ++n) er[n] = (half_t)0.0f;
}

// message pass: gather phi[n1]*w_s, scatter-add to s[n0] / v_new[n0]
__global__ void k_message(const int* __restrict__ nbrs, const float* __restrict__ phi,
                          const float* __restrict__ ws, const float* __restrict__ unit,
                          const float* __restrict__ vold, float* __restrict__ s,
                          float* __restrict__ vnew) {
  int i = blockIdx.x * blockDim.x + threadIdx.x;
  if (i >= NEDGES * FEAT) return;
  int e = i >> 7, f = i & 127;
  int n0 = nbrs[2 * e], n1 = nbrs[2 * e + 1];
  const float* wr = ws + (size_t)e * 3 * FEAT;
  const float* pr = phi + (size_t)n1 * 3 * FEAT;
  float inv0 = pr[f] * wr[f];
  float dsij = pr[FEAT + f] * wr[FEAT + f];
  float inv2 = pr[2 * FEAT + f] * wr[2 * FEAT + f];
  atomAddF(&s[(size_t)n0 * FEAT + f], dsij);
#pragma unroll
  for (int c = 0; c < 3; ++c) {
    float dv = inv2 * unit[e * 3 + c] + inv0 * vold[((size_t)n1 * 3 + c) * FEAT + f];
    atomAddF(&vnew[((size_t)n0 * 3 + c) * FEAT + f], dv);
  }
}

// cat = [s, norm(v_v)] as f16 (node update MLP input)
__global__ void k_cat_node(const float* __restrict__ s, const float* __restrict__ vv,
                           half_t* __restrict__ cat) {
  int i = blockIdx.x * blockDim.x + threadIdx.x;
  if (i >= NATOMS * FEAT) return;
  int n = i >> 7, g = i & 127;
  float a = vv[((size_t)n * 3 + 0) * FEAT + g];
  float b = vv[((size_t)n * 3 + 1) * FEAT + g];
  float c = vv[((size_t)n * 3 + 2) * FEAT + g];
  cat[(size_t)n * 2 * FEAT + g]        = (half_t)s[i];
  cat[(size_t)n * 2 * FEAT + FEAT + g] = (half_t)sqrtf(a * a + b * b + c * c + EPS3);
}

__global__ void k_update(const float* __restrict__ u, const float* __restrict__ vv,
                         const float* __restrict__ a, float* __restrict__ s,
                         float* __restrict__ v) {
  int i = blockIdx.x * blockDim.x + threadIdx.x;
  if (i >= NATOMS * FEAT) return;
  int n = i >> 7, g = i & 127;
  const float* ar = a + (size_t)n * 3 * FEAT;
  float a0 = ar[g], a1 = ar[FEAT + g], a2 = ar[2 * FEAT + g];
  float dot = 0.0f;
#pragma unroll
  for (int c = 0; c < 3; ++c) {
    size_t idx = ((size_t)n * 3 + c) * FEAT + g;
    float uc = u[idx];
    dot += uc * vv[idx];
    v[idx] += uc * a0;
  }
  s[i] += dot * a1 + a2;
}

__global__ void k_norm_node(const float* __restrict__ vv, float* __restrict__ vn) {
  int i = blockIdx.x * blockDim.x + threadIdx.x;
  if (i >= NATOMS * FEAT) return;
  int n = i >> 7, g = i & 127;
  float a = vv[((size_t)n * 3 + 0) * FEAT + g];
  float b = vv[((size_t)n * 3 + 1) * FEAT + g];
  float c = vv[((size_t)n * 3 + 2) * FEAT + g];
  vn[i] = sqrtf(a * a + b * b + c * c + EPS3);
}

__global__ void k_sprod(const int* __restrict__ nbrs, const float* __restrict__ s,
                        half_t* __restrict__ sp) {
  int i = blockIdx.x * blockDim.x + threadIdx.x;
  if (i >= NEDGES * FEAT) return;
  int e = i >> 7, f = i & 127;
  int n0 = nbrs[2 * e], n1 = nbrs[2 * e + 1];
  sp[i] = (half_t)(s[(size_t)n0 * FEAT + f] * s[(size_t)n1 * FEAT + f]);
}

__global__ void k_cat_edge(const int* __restrict__ nbrs, const float* __restrict__ s,
                           const float* __restrict__ vn, half_t* __restrict__ cat) {
  int i = blockIdx.x * blockDim.x + threadIdx.x;
  if (i >= NEDGES * FEAT) return;
  int e = i >> 7, g = i & 127;
  int n0 = nbrs[2 * e], n1 = nbrs[2 * e + 1];
  cat[(size_t)e * 2 * FEAT + g]        = (half_t)s[(size_t)n0 * FEAT + g];
  cat[(size_t)e * 2 * FEAT + FEAT + g] = (half_t)vn[(size_t)n1 * FEAT + g];
}

__global__ void k_prdot(const int* __restrict__ nbrs, const float* __restrict__ s,
                        const float* __restrict__ v, const float* __restrict__ unit,
                        half_t* __restrict__ pin) {
  int i = blockIdx.x * blockDim.x + threadIdx.x;
  if (i >= NEDGES * FEAT) return;
  int e = i >> 7, f = i & 127;
  int n0 = nbrs[2 * e], n1 = nbrs[2 * e + 1];
  float pr = v[((size_t)n1 * 3 + 0) * FEAT + f] * unit[e * 3 + 0]
           + v[((size_t)n1 * 3 + 1) * FEAT + f] * unit[e * 3 + 1]
           + v[((size_t)n1 * 3 + 2) * FEAT + f] * unit[e * 3 + 2];
  pin[i] = (half_t)(s[(size_t)n0 * FEAT + f] * pr);
}

// wave-per-edge: f_edge += (qq_scal + qp2_scal)*unit + f_qp1
__global__ __launch_bounds__(256) void k_force(
    const int* __restrict__ nbrs,
    const half_t* __restrict__ hq,  const float* __restrict__ qqW2,  const float* __restrict__ qqb2,
    const half_t* __restrict__ hq2, const float* __restrict__ qp2W2, const float* __restrict__ qp2b2,
    const float* __restrict__ u, const float* __restrict__ avv,
    const float* __restrict__ unit, float* __restrict__ fe)
{
  int e = (int)(blockIdx.x * blockDim.x + threadIdx.x) >> 5;
  if (e >= NEDGES) return;
  int lane = threadIdx.x & 31;
  int n1 = nbrs[2 * e + 1];
  float d1 = 0.0f, d2 = 0.0f;
#pragma unroll
  for (int j = 0; j < 2; ++j) {
    int f = 2 * lane + j;
    d1 += (float)hq[(size_t)e * 64 + f] * qqW2[f];
    d2 += (float)hq2[(size_t)e * 64 + f] * qp2W2[f];
  }
  float fq0 = 0.0f, fq1 = 0.0f, fq2 = 0.0f;
#pragma unroll
  for (int j = 0; j < 4; ++j) {
    int f = 4 * lane + j;
    float av = avv[(size_t)e * FEAT + f];
    fq0 += u[((size_t)n1 * 3 + 0) * FEAT + f] * av;
    fq1 += u[((size_t)n1 * 3 + 1) * FEAT + f] * av;
    fq2 += u[((size_t)n1 * 3 + 2) * FEAT + f] * av;
  }
#pragma unroll
  for (int off = 16; off > 0; off >>= 1) {
    d1  += __shfl_xor(d1, off, 32);
    d2  += __shfl_xor(d2, off, 32);
    fq0 += __shfl_xor(fq0, off, 32);
    fq1 += __shfl_xor(fq1, off, 32);
    fq2 += __shfl_xor(fq2, off, 32);
  }
  if (lane < 3) {
    float fq = (lane == 0) ? fq0 : (lane == 1) ? fq1 : fq2;
    float sc = (d1 + qqb2[0]) + (d2 + qp2b2[0]);
    fe[(size_t)e * 3 + lane] += fq + sc * unit[e * 3 + lane];
  }
}

__global__ void k_scatter(const int* __restrict__ nbrs, const float* __restrict__ fe,
                          float* __restrict__ out) {
  int i = blockIdx.x * blockDim.x + threadIdx.x;
  if (i >= NEDGES * 3) return;
  int e = i / 3, c = i - e * 3;
  atomAddF(&out[nbrs[2 * e] * 3 + c], fe[i]);
}

// ---------------------------------------------------------------------------
// Host orchestration
// ---------------------------------------------------------------------------
extern "C" void kernel_launch(void* const* d_in, const int* in_sizes, int n_in,
                              void* d_out, int out_size, void* d_ws, size_t ws_size,
                              hipStream_t stream)
{
  (void)in_sizes; (void)n_in; (void)out_size; (void)ws_size;
  const float* nxyz = (const float*)d_in[0];
  const int*   nbrs = (const int*)d_in[1];
  const float* emb  = (const float*)d_in[2];
  enum { MSG_W1 = 0, MSG_W2, MSG_WD, MSG_B1, MSG_B2, MSG_BD,
         QP1_U, QP1_V, QP1_W1, QP1_W2, QP1_B1, QP1_B2,
         QP2_W1, QP2_W2, QP2_B1, QP2_B2,
         QQ_W1, QQ_W2, QQ_B1, QQ_B2,
         UPD_U, UPD_V, UPD_W1, UPD_W2, UPD_B1, UPD_B2 };
  auto P = [&](int L, int k) -> const float* { return (const float*)d_in[3 + L * 26 + k]; };

  // bump arena over d_ws
  char* base = (char*)d_ws; size_t off = 0;
  auto alloc = [&](size_t bytes) -> void* {
    off = (off + 255) & ~(size_t)255;
    void* p = base + off; off += bytes; return p;
  };
  const int N = NATOMS, E = NEDGES, F = FEAT;

  float*  s    = (float*) alloc((size_t)N * F * 4);
  half_t* sh   = (half_t*)alloc((size_t)N * F * 2);
  float*  vA   = (float*) alloc((size_t)N * 3 * F * 4);
  float*  vB   = (float*) alloc((size_t)N * 3 * F * 4);
  half_t* vh   = (half_t*)alloc((size_t)N * 3 * F * 2);
  float*  unit = (float*) alloc((size_t)E * 3 * 4);
  float*  env  = (float*) alloc((size_t)E * 4);
  half_t* erbf = (half_t*)alloc((size_t)E * RBFPAD * 2);
  float*  fe   = (float*) alloc((size_t)E * 3 * 4);
  float*  phi  = (float*) alloc((size_t)N * 3 * F * 4);
  half_t* hn   = (half_t*)alloc((size_t)N * F * 2);
  half_t* catn = (half_t*)alloc((size_t)N * 2 * F * 2);
  float*  an   = (float*) alloc((size_t)N * 3 * F * 4);
  float*  uv   = (float*) alloc((size_t)N * 3 * F * 4);
  float*  vvb  = (float*) alloc((size_t)N * 3 * F * 4);
  float*  vnn  = (float*) alloc((size_t)N * F * 4);
  float*  wsb  = (float*) alloc((size_t)E * 3 * F * 4);   // 245 MB
  half_t* sp   = (half_t*)alloc((size_t)E * F * 2);
  half_t* hq   = (half_t*)alloc((size_t)E * 64 * 2);
  half_t* cate = (half_t*)alloc((size_t)E * 2 * F * 2);
  half_t* hqp1 = (half_t*)alloc((size_t)E * F * 2);
  float*  avv  = (float*) alloc((size_t)E * F * 4);
  half_t* pin  = (half_t*)alloc((size_t)E * F * 2);
  half_t* hq2  = (half_t*)alloc((size_t)E * 64 * 2);

  // f16 transposed weights (all layers, converted every call — tiny)
  half_t *wMsg1[3], *wMsg2[3], *wMsgD[3], *wUpdU[3], *wUpdV[3], *wUpd1[3], *wUpd2[3];
  half_t *wQq1[3], *wQp1U[3], *wQp1V[3], *wQp11[3], *wQp12[3], *wQp21[3];
  for (int L = 0; L < 3; ++L) {
    struct { half_t** dst; int pk, Kv, Kp, Nn; } cv[13] = {
      {&wMsg1[L], MSG_W1, 128, 128, 128}, {&wMsg2[L], MSG_W2, 128, 128, 384},
      {&wMsgD[L], MSG_WD, NRBF, RBFPAD, 384},
      {&wUpdU[L], UPD_U, 128, 128, 128}, {&wUpdV[L], UPD_V, 128, 128, 128},
      {&wUpd1[L], UPD_W1, 256, 256, 128}, {&wUpd2[L], UPD_W2, 128, 128, 384},
      {&wQq1[L],  QQ_W1, 128, 128, 64},
      {&wQp1U[L], QP1_U, 128, 128, 128}, {&wQp1V[L], QP1_V, 128, 128, 128},
      {&wQp11[L], QP1_W1, 256, 256, 128}, {&wQp12[L], QP1_W2, 128, 128, 128},
      {&wQp21[L], QP2_W1, 128, 128, 64},
    };
    for (int k = 0; k < 13; ++k) {
      int n = cv[k].Kp * cv[k].Nn;
      *cv[k].dst = (half_t*)alloc((size_t)n * 2);
      k_wpadT<<<CEILDIV(n, 256), 256, 0, stream>>>(*cv[k].dst, P(L, cv[k].pk),
                                                   cv[k].Kv, cv[k].Kp, cv[k].Nn);
    }
  }

  // ---- init ----
  k_zero<<<CEILDIV(N * 3 * F, 256), 256, 0, stream>>>(vA, N * 3 * F);
  k_zero<<<CEILDIV(E * 3, 256), 256, 0, stream>>>(fe, E * 3);
  k_zero<<<CEILDIV(N * 3, 256), 256, 0, stream>>>((float*)d_out, N * 3);
  k_embed<<<CEILDIV(N * F, 256), 256, 0, stream>>>(nxyz, emb, s, sh);
  k_geom<<<CEILDIV(E, 256), 256, 0, stream>>>(nxyz, nbrs, unit, env, erbf);

  constexpr int BS_H = GF_BIAS | GF_SILU | GF_OF16;  // hidden MLP layers
  constexpr int B_F  = GF_BIAS | GF_OF32;            // output MLP layers
  constexpr int BR_F = GF_BIAS | GF_RS | GF_OF32;    // rbf filter (*env)
  constexpr int P_F  = GF_OF32;                      // plain einsum GEMMs

  float* vcur = vA; float* vnxt = vB;
  for (int L = 0; L < 3; ++L) {
    if (L > 0) k_f2h<<<CEILDIV(N * F, 256), 256, 0, stream>>>(sh, s, N * F);

    // phi = mlp2(s)                                  [node GEMMs]
    gemm_t<128, BS_H>(stream, sh, wMsg1[L], P(L, MSG_B1), nullptr, nullptr, hn, N, F);
    gemm_t<128, B_F >(stream, hn, wMsg2[L], P(L, MSG_B2), nullptr, phi, nullptr, N, 3 * F);
    // w_s = (rbf @ Wd + bd) * env                    [edge GEMM, K padded 20->32]
    gemm_t<32,  BR_F>(stream, erbf, wMsgD[L], P(L, MSG_BD), env, wsb, nullptr, E, 3 * F);

    // message scatter (double-buffered v)
    hipMemcpyAsync(vnxt, vcur, (size_t)N * 3 * F * 4, hipMemcpyDeviceToDevice, stream);
    k_message<<<CEILDIV(E * F, 256), 256, 0, stream>>>(nbrs, phi, wsb, unit, vcur, s, vnxt);
    { float* t = vcur; vcur = vnxt; vnxt = t; }

    // node update block
    k_f2h<<<CEILDIV(N * 3 * F, 256), 256, 0, stream>>>(vh, vcur, N * 3 * F);
    gemm_t<128, P_F >(stream, vh, wUpdU[L], nullptr, nullptr, uv, nullptr, 3 * N, F);
    gemm_t<128, P_F >(stream, vh, wUpdV[L], nullptr, nullptr, vvb, nullptr, 3 * N, F);
    k_cat_node<<<CEILDIV(N * F, 256), 256, 0, stream>>>(s, vvb, catn);
    gemm_t<256, BS_H>(stream, catn, wUpd1[L], P(L, UPD_B1), nullptr, nullptr, hn, N, F);
    gemm_t<128, B_F >(stream, hn, wUpd2[L], P(L, UPD_B2), nullptr, an, nullptr, N, 3 * F);
    k_update<<<CEILDIV(N * F, 256), 256, 0, stream>>>(uv, vvb, an, s, vcur);

    // qp1 einsums hoisted to node level: u_vn = v@U, vv_n = v@V, norms per node
    k_f2h<<<CEILDIV(N * 3 * F, 256), 256, 0, stream>>>(vh, vcur, N * 3 * F);
    gemm_t<128, P_F >(stream, vh, wQp1U[L], nullptr, nullptr, uv, nullptr, 3 * N, F);
    gemm_t<128, P_F >(stream, vh, wQp1V[L], nullptr, nullptr, vvb, nullptr, 3 * N, F);
    k_norm_node<<<CEILDIV(N * F, 256), 256, 0, stream>>>(vvb, vnn);

    // edge MLPs
    k_sprod<<<CEILDIV(E * F, 256), 256, 0, stream>>>(nbrs, s, sp);
    gemm_t<128, BS_H>(stream, sp, wQq1[L], P(L, QQ_B1), nullptr, nullptr, hq, E, 64);
    k_cat_edge<<<CEILDIV(E * F, 256), 256, 0, stream>>>(nbrs, s, vnn, cate);
    gemm_t<256, BS_H>(stream, cate, wQp11[L], P(L, QP1_B1), nullptr, nullptr, hqp1, E, F);
    gemm_t<128, B_F >(stream, hqp1, wQp12[L], P(L, QP1_B2), nullptr, avv, nullptr, E, F);
    k_prdot<<<CEILDIV(E * F, 256), 256, 0, stream>>>(nbrs, s, vcur, unit, pin);
    gemm_t<128, BS_H>(stream, pin, wQp21[L], P(L, QP2_B1), nullptr, nullptr, hq2, E, 64);

    // per-edge force accumulation (wave per edge)
    k_force<<<CEILDIV(E * 32, 256), 256, 0, stream>>>(
        nbrs, hq, P(L, QQ_W2), P(L, QQ_B2), hq2, P(L, QP2_W2), P(L, QP2_B2),
        uv, avv, unit, fe);
  }

  k_scatter<<<CEILDIV(E * 3, 256), 256, 0, stream>>>(nbrs, fe, (float*)d_out);
}
